// PatientLevelMultiTaskModel_10831907520633
// MI455X (gfx1250) — matrix operand order repaired and checked by reading, not compiled, not run on hardware
//
#include <hip/hip_runtime.h>
#include <cstdint>

// ---------------------------------------------------------------------------
// PatientLevelMultiTaskModel forward for MI455X (gfx1250, wave32, WMMA)
// fp32 in/out, f16 WMMA compute with fp32 accumulate on the big GEMMs.
// ---------------------------------------------------------------------------

typedef __attribute__((ext_vector_type(16))) _Float16 v16h;
typedef __attribute__((ext_vector_type(8)))  _Float16 v8h;
typedef __attribute__((ext_vector_type(8)))  float    v8f;

#define LDSK 40   // halves per LDS tile row: 32 data + 8 pad = 80 bytes (16B aligned)

static constexpr int Bb   = 8;
static constexpr int Gg   = 256;
static constexpr int Tt   = 2048;
static constexpr int Hh   = 8;
static constexpr int Dg   = 768;    // GLOM_DIM
static constexpr int Dt   = 1536;   // TILE_DIM
static constexpr int Dh   = 96;     // head dim
static constexpr float LN_EPS = 1e-5f;

__device__ __forceinline__ void pack16(v16h& f, const v8h a, const v8h b) {
  #pragma unroll
  for (int i = 0; i < 8; ++i) { f[i] = a[i]; f[i + 8] = b[i]; }
}

// ---------------------------------------------------------------------------
// WMMA GEMM: C = alpha * (A[MxK] * W[NxK]^T) + bias[N]  (+C if accumulate)
// Block: 256 threads = 8 waves (2x4), block tile 64(M) x 128(N), K-step 32.
// Wave tile 32x32 -> 4 WMMAs per wave per K-step, LDS double-buffered,
// one barrier per K-step, branch-free clamped staging. Requires K % 32 == 0.
// Batched via blockIdx.z: ptr += (z>>zshift)*s0 + (z&mask)*s1.
// ---------------------------------------------------------------------------
__global__ void __launch_bounds__(256) gemm_abt_wmma(
    const float* __restrict__ A, const float* __restrict__ W,
    const float* __restrict__ bias, float* __restrict__ C,
    int M, int N, int K, int lda, int ldw, int ldc,
    long sA0, long sA1, long sW0, long sW1, long sC0, long sC1, int zshift,
    float alpha, int accumulate, int act)
{
  __shared__ _Float16 Ah[2][64 * LDSK];
  __shared__ _Float16 Wh[2][128 * LDSK];

  int z  = blockIdx.z;
  int zo = z >> zshift, zi = z & ((1 << zshift) - 1);
  A += zo * sA0 + zi * sA1;
  W += zo * sW0 + zi * sW1;
  C += zo * sC0 + zi * sC1;

  int tid  = threadIdx.x;
  int wave = tid >> 5, lane = tid & 31;
  int wm = wave >> 2, wn = wave & 3;      // 2 x 4 waves, wave tile 32x32
  int l  = lane & 15, hi = lane >> 4;
  int m0 = blockIdx.x * 64, n0 = blockIdx.y * 128;

  // k-invariant staging maps with clamped rows (branch-free bounds)
  long aoff[2]; bool av[2]; int arow[2], ac4[2];
  #pragma unroll
  for (int j = 0; j < 2; ++j) {
    int idx = tid + j * 256;
    int row = idx >> 3, c4 = (idx & 7) << 2;
    int gr  = m0 + row;
    av[j]   = gr < M;
    int cr  = av[j] ? gr : (M - 1);
    aoff[j] = (long)cr * lda + c4;
    arow[j] = row; ac4[j] = c4;
  }
  long boff[4]; bool bv[4]; int brow[4], bc4[4];
  #pragma unroll
  for (int j = 0; j < 4; ++j) {
    int idx = tid + j * 256;
    int row = idx >> 3, c4 = (idx & 7) << 2;
    int gr  = n0 + row;
    bv[j]   = gr < N;
    int cr  = bv[j] ? gr : (N - 1);
    boff[j] = (long)cr * ldw + c4;
    brow[j] = row; bc4[j] = c4;
  }

  v8f acc[4] = {{}, {}, {}, {}};
  float4 ra[2], rb[4];
  const int nk = K >> 5;

  // prologue: fetch K-tile 0 into registers
  #pragma unroll
  for (int j = 0; j < 2; ++j) ra[j] = *(const float4*)(A + aoff[j]);
  #pragma unroll
  for (int j = 0; j < 4; ++j) rb[j] = *(const float4*)(W + boff[j]);

  for (int t = 0; t < nk; ++t) {
    int cur = t & 1;
    // stage registers (tile t) -> LDS buffer `cur`, fp32 -> f16, masked
    #pragma unroll
    for (int j = 0; j < 2; ++j) {
      _Float16* d = Ah[cur] + arow[j] * LDSK + ac4[j];
      d[0] = av[j] ? (_Float16)ra[j].x : (_Float16)0.f;
      d[1] = av[j] ? (_Float16)ra[j].y : (_Float16)0.f;
      d[2] = av[j] ? (_Float16)ra[j].z : (_Float16)0.f;
      d[3] = av[j] ? (_Float16)ra[j].w : (_Float16)0.f;
    }
    #pragma unroll
    for (int j = 0; j < 4; ++j) {
      _Float16* d = Wh[cur] + brow[j] * LDSK + bc4[j];
      d[0] = bv[j] ? (_Float16)rb[j].x : (_Float16)0.f;
      d[1] = bv[j] ? (_Float16)rb[j].y : (_Float16)0.f;
      d[2] = bv[j] ? (_Float16)rb[j].z : (_Float16)0.f;
      d[3] = bv[j] ? (_Float16)rb[j].w : (_Float16)0.f;
    }
    // issue global loads for tile t+1 (consumed next iteration, after WMMAs)
    if (t + 1 < nk) {
      int k0 = (t + 1) << 5;
      #pragma unroll
      for (int j = 0; j < 2; ++j) ra[j] = *(const float4*)(A + aoff[j] + k0);
      #pragma unroll
      for (int j = 0; j < 4; ++j) rb[j] = *(const float4*)(W + boff[j] + k0);
    }
    if (t + 2 < nk) {
      __builtin_prefetch(A + aoff[0] + ((t + 2) << 5), 0, 1);
      __builtin_prefetch(W + boff[0] + ((t + 2) << 5), 0, 1);
    }
    __syncthreads();

    // fragments from LDS buffer `cur`
    v16h af[2], bf[2];
    #pragma unroll
    for (int i = 0; i < 2; ++i) {
      const _Float16* pa = Ah[cur] + (wm * 32 + i * 16 + l) * LDSK;
      pack16(af[i], *(const v8h*)(pa + hi * 16),
                    *(const v8h*)(pa + hi * 16 + 8));
    }
    #pragma unroll
    for (int i = 0; i < 2; ++i) {
      const _Float16* pb = Wh[cur] + (wn * 32 + i * 16 + l) * LDSK + hi * 16;
      pack16(bf[i], *(const v8h*)(pb), *(const v8h*)(pb + 8));
    }
    acc[0] = __builtin_amdgcn_wmma_f32_16x16x32_f16(false, af[0], false, bf[0], (short)0, acc[0], false, false);
    acc[1] = __builtin_amdgcn_wmma_f32_16x16x32_f16(false, af[0], false, bf[1], (short)0, acc[1], false, false);
    acc[2] = __builtin_amdgcn_wmma_f32_16x16x32_f16(false, af[1], false, bf[0], (short)0, acc[2], false, false);
    acc[3] = __builtin_amdgcn_wmma_f32_16x16x32_f16(false, af[1], false, bf[1], (short)0, acc[3], false, false);
  }

  // epilogue
  #pragma unroll
  for (int bi = 0; bi < 2; ++bi) {
    int col = n0 + wn * 32 + bi * 16 + l;
    if (col >= N) continue;
    float bvv = bias ? bias[col] : 0.f;
    #pragma unroll
    for (int ai = 0; ai < 2; ++ai) {
      v8f a = acc[ai * 2 + bi];
      #pragma unroll
      for (int r = 0; r < 8; ++r) {
        int row = m0 + wm * 32 + ai * 16 + r + hi * 8;
        if (row < M) {
          long off = (long)row * ldc + col;
          float v = a[r] * alpha + bvv;
          if (accumulate) v += C[off];
          if (act == 1) v = v > 0.f ? v : 0.f;
          C[off] = v;
        }
      }
    }
  }
}

// ---------------------------------------------------------------------------
// WMMA GEMM: C = A[MxK] * B[KxN]  (B row-major, staged transposed into LDS)
// Same blocking as above. Requires K % 32 == 0.
// ---------------------------------------------------------------------------
__global__ void __launch_bounds__(256) gemm_ab_wmma(
    const float* __restrict__ A, const float* __restrict__ Bm,
    float* __restrict__ C,
    int M, int N, int K, int lda, int ldb, int ldc,
    long sA0, long sA1, long sB0, long sB1, long sC0, long sC1, int zshift)
{
  __shared__ _Float16 Ah[2][64 * LDSK];
  __shared__ _Float16 Wh[2][128 * LDSK];   // B^T tile: [n][k]

  int z  = blockIdx.z;
  int zo = z >> zshift, zi = z & ((1 << zshift) - 1);
  A  += zo * sA0 + zi * sA1;
  Bm += zo * sB0 + zi * sB1;
  C  += zo * sC0 + zi * sC1;

  int tid  = threadIdx.x;
  int wave = tid >> 5, lane = tid & 31;
  int wm = wave >> 2, wn = wave & 3;
  int l  = lane & 15, hi = lane >> 4;
  int m0 = blockIdx.x * 64, n0 = blockIdx.y * 128;

  long aoff[2]; bool av[2]; int arow[2], ac4[2];
  #pragma unroll
  for (int j = 0; j < 2; ++j) {
    int idx = tid + j * 256;
    int row = idx >> 3, c4 = (idx & 7) << 2;
    int gr  = m0 + row;
    av[j]   = gr < M;
    int cr  = av[j] ? gr : (M - 1);
    aoff[j] = (long)cr * lda + c4;
    arow[j] = row; ac4[j] = c4;
  }
  // B tile 32(k) x 128(n): thread j covers (kr, n4); N assumed multiple of 4
  long boff[4]; bool bv[4]; int bkr[4], bn4[4];
  #pragma unroll
  for (int j = 0; j < 4; ++j) {
    int idx = tid + j * 256;               // 0..1023 over 32x32 float4s
    int kr = idx >> 5, n4 = (idx & 31) << 2;
    int gn = n0 + n4;
    bv[j]  = gn < N;
    int cn = bv[j] ? gn : (N - 4);
    boff[j] = (long)kr * ldb + cn;
    bkr[j] = kr; bn4[j] = n4;
  }

  v8f acc[4] = {{}, {}, {}, {}};
  float4 ra[2], rb[4];
  const int nk = K >> 5;

  #pragma unroll
  for (int j = 0; j < 2; ++j) ra[j] = *(const float4*)(A + aoff[j]);
  #pragma unroll
  for (int j = 0; j < 4; ++j) rb[j] = *(const float4*)(Bm + boff[j]);

  for (int t = 0; t < nk; ++t) {
    int cur = t & 1;
    #pragma unroll
    for (int j = 0; j < 2; ++j) {
      _Float16* d = Ah[cur] + arow[j] * LDSK + ac4[j];
      d[0] = av[j] ? (_Float16)ra[j].x : (_Float16)0.f;
      d[1] = av[j] ? (_Float16)ra[j].y : (_Float16)0.f;
      d[2] = av[j] ? (_Float16)ra[j].z : (_Float16)0.f;
      d[3] = av[j] ? (_Float16)ra[j].w : (_Float16)0.f;
    }
    #pragma unroll
    for (int j = 0; j < 4; ++j) {   // transposed store: [n][k]
      _Float16* d = Wh[cur] + bn4[j] * LDSK + bkr[j];
      d[0 * LDSK] = bv[j] ? (_Float16)rb[j].x : (_Float16)0.f;
      d[1 * LDSK] = bv[j] ? (_Float16)rb[j].y : (_Float16)0.f;
      d[2 * LDSK] = bv[j] ? (_Float16)rb[j].z : (_Float16)0.f;
      d[3 * LDSK] = bv[j] ? (_Float16)rb[j].w : (_Float16)0.f;
    }
    if (t + 1 < nk) {
      long k0  = (long)(t + 1) << 5;
      long kb0 = k0 * ldb;
      #pragma unroll
      for (int j = 0; j < 2; ++j) ra[j] = *(const float4*)(A + aoff[j] + k0);
      #pragma unroll
      for (int j = 0; j < 4; ++j) rb[j] = *(const float4*)(Bm + boff[j] + kb0);
    }
    if (t + 2 < nk) {
      __builtin_prefetch(A + aoff[0] + ((long)(t + 2) << 5), 0, 1);
      __builtin_prefetch(Bm + boff[0] + ((long)(t + 2) << 5) * ldb, 0, 1);
    }
    __syncthreads();

    v16h af[2], bf[2];
    #pragma unroll
    for (int i = 0; i < 2; ++i) {
      const _Float16* pa = Ah[cur] + (wm * 32 + i * 16 + l) * LDSK;
      pack16(af[i], *(const v8h*)(pa + hi * 16),
                    *(const v8h*)(pa + hi * 16 + 8));
    }
    #pragma unroll
    for (int i = 0; i < 2; ++i) {
      const _Float16* pb = Wh[cur] + (wn * 32 + i * 16 + l) * LDSK + hi * 16;
      pack16(bf[i], *(const v8h*)(pb), *(const v8h*)(pb + 8));
    }
    acc[0] = __builtin_amdgcn_wmma_f32_16x16x32_f16(false, af[0], false, bf[0], (short)0, acc[0], false, false);
    acc[1] = __builtin_amdgcn_wmma_f32_16x16x32_f16(false, af[0], false, bf[1], (short)0, acc[1], false, false);
    acc[2] = __builtin_amdgcn_wmma_f32_16x16x32_f16(false, af[1], false, bf[0], (short)0, acc[2], false, false);
    acc[3] = __builtin_amdgcn_wmma_f32_16x16x32_f16(false, af[1], false, bf[1], (short)0, acc[3], false, false);
  }

  #pragma unroll
  for (int bi = 0; bi < 2; ++bi) {
    int col = n0 + wn * 32 + bi * 16 + l;
    if (col >= N) continue;
    #pragma unroll
    for (int ai = 0; ai < 2; ++ai) {
      v8f a = acc[ai * 2 + bi];
      #pragma unroll
      for (int r = 0; r < 8; ++r) {
        int row = m0 + wm * 32 + ai * 16 + r + hi * 8;
        if (row < M) C[(long)row * ldc + col] = a[r];
      }
    }
  }
}

// ---------------------------------------------------------------------------
// Helpers / elementwise kernels
// ---------------------------------------------------------------------------
__device__ __forceinline__ float block_sum256(float v, float* red) {
  #pragma unroll
  for (int o = 16; o > 0; o >>= 1) v += __shfl_xor(v, o, 32);
  int tid = threadIdx.x;
  if ((tid & 31) == 0) red[tid >> 5] = v;
  __syncthreads();
  float t = 0.f;
  #pragma unroll
  for (int i = 0; i < 8; ++i) t += red[i];
  __syncthreads();
  return t;
}

__device__ __forceinline__ float block_max256(float v, float* red) {
  #pragma unroll
  for (int o = 16; o > 0; o >>= 1) v = fmaxf(v, __shfl_xor(v, o, 32));
  int tid = threadIdx.x;
  if ((tid & 31) == 0) red[tid >> 5] = v;
  __syncthreads();
  float t = -INFINITY;
  #pragma unroll
  for (int i = 0; i < 8; ++i) t = fmaxf(t, red[i]);
  __syncthreads();
  return t;
}

// pos1: out[r,j] = relu(w[j,0]*c0 + w[j,1]*c1 + b[j]), 16384 x 256
__global__ void __launch_bounds__(256) pos1_kernel(
    const float* __restrict__ coords, const float* __restrict__ w,
    const float* __restrict__ b, float* __restrict__ out)
{
  long i = (long)blockIdx.x * 256 + threadIdx.x;
  int row = (int)(i >> 8);
  int j   = (int)(i & 255);
  float c0 = coords[row * 2 + 0], c1 = coords[row * 2 + 1];
  float v = w[j * 2 + 0] * c0 + w[j * 2 + 1] * c1 + b[j];
  out[i] = v > 0.f ? v : 0.f;
}

// LayerNorm over last dim D with optional residual and activation.
// act: 0=none, 1=relu.  In-place safe (out may alias x).
__global__ void __launch_bounds__(256) ln_kernel(
    const float* __restrict__ X, const float* __restrict__ R,
    const float* __restrict__ gam, const float* __restrict__ bet,
    float* __restrict__ O, int D, int act)
{
  __shared__ float red[8];
  long row = blockIdx.x;
  const float* x = X + row * D;
  const float* r = R ? R + row * D : nullptr;
  float* o = O + row * D;
  int tid = threadIdx.x;

  float s = 0.f;
  for (int i = tid; i < D; i += 256) s += x[i] + (r ? r[i] : 0.f);
  float mean = block_sum256(s, red) / (float)D;

  float vs = 0.f;
  for (int i = tid; i < D; i += 256) {
    float v = x[i] + (r ? r[i] : 0.f) - mean;
    vs += v * v;
  }
  float var  = block_sum256(vs, red) / (float)D;
  float rstd = rsqrtf(var + LN_EPS);

  for (int i = tid; i < D; i += 256) {
    float v = (x[i] + (r ? r[i] : 0.f) - mean) * rstd * gam[i] + bet[i];
    if (act == 1) v = v > 0.f ? v : 0.f;
    o[i] = v;
  }
}

// Masked row softmax over T keys; mask[b*T + t] != 0 -> -inf.
__global__ void __launch_bounds__(256) softmax_mask_kernel(
    float* __restrict__ S, const unsigned char* __restrict__ mask,
    int T, int rows_per_b)
{
  __shared__ float red[8];
  long row = blockIdx.x;
  int b = (int)(row / rows_per_b);
  float* p = S + row * T;
  const unsigned char* mk = mask + (long)b * T;
  int tid = threadIdx.x;

  float mx = -INFINITY;
  for (int t = tid; t < T; t += 256) {
    float v = mk[t] ? -INFINITY : p[t];
    mx = fmaxf(mx, v);
  }
  mx = block_max256(mx, red);

  float s = 0.f;
  for (int t = tid; t < T; t += 256) {
    float e = mk[t] ? 0.f : __expf(p[t] - mx);
    p[t] = e;
    s += e;
  }
  s = block_sum256(s, red);
  float inv = 1.f / s;
  for (int t = tid; t < T; t += 256) p[t] *= inv;
}

// Generic small dense: O[m,n] = act(sum_k X[m,k]*W[n,k] + b[n])
// act: 0=none, 1=relu, 2=sigmoid
__global__ void __launch_bounds__(256) linear_small_kernel(
    const float* __restrict__ X, const float* __restrict__ W,
    const float* __restrict__ b, float* __restrict__ O,
    int M, int N, int K, int act)
{
  long i = (long)blockIdx.x * 256 + threadIdx.x;
  if (i >= (long)M * N) return;
  int m = (int)(i / N), n = (int)(i % N);
  const float* x = X + (long)m * K;
  const float* w = W + (long)n * K;
  float s = b ? b[n] : 0.f;
  for (int k = 0; k < K; ++k) s += x[k] * w[k];
  if (act == 1) s = s > 0.f ? s : 0.f;
  else if (act == 2) s = 1.f / (1.f + __expf(-s));
  O[i] = s;
}

// glom <- lesion cross attention (2 keys, per-head), wave per (bg,h)
__global__ void __launch_bounds__(32) les_attn_kernel(
    const float* __restrict__ Q, const float* __restrict__ Kl,
    const float* __restrict__ Vl, float* __restrict__ O)
{
  int bgh = blockIdx.x;             // (B*G)*H
  int h = bgh & 7, bg = bgh >> 3;
  int t = threadIdx.x;
  const float* q  = Q  + (long)bg * Dg + h * Dh;
  const float* k0 = Kl + h * Dh;
  const float* k1 = Kl + Dg + h * Dh;
  const float* v0 = Vl + h * Dh;
  const float* v1 = Vl + Dg + h * Dh;
  float s0 = 0.f, s1 = 0.f;
  #pragma unroll
  for (int j = 0; j < 3; ++j) {
    int d = t + 32 * j;
    float qq = q[d];
    s0 += qq * k0[d];
    s1 += qq * k1[d];
  }
  #pragma unroll
  for (int o = 16; o > 0; o >>= 1) {
    s0 += __shfl_xor(s0, o, 32);
    s1 += __shfl_xor(s1, o, 32);
  }
  const float sc = 0.1020620726f;   // 1/sqrt(96)
  s0 *= sc; s1 *= sc;
  float m  = fmaxf(s0, s1);
  float e0 = __expf(s0 - m), e1 = __expf(s1 - m);
  float inv = 1.f / (e0 + e1);
  e0 *= inv; e1 *= inv;
  float* o = O + (long)bg * Dg + h * Dh;
  #pragma unroll
  for (int j = 0; j < 3; ++j) {
    int d = t + 32 * j;
    o[d] = e0 * v0[d] + e1 * v1[d];
  }
}

// masked mean over G: O[b,d] = sum_{g valid} X[b,g,d] / count(valid)
__global__ void __launch_bounds__(256) masked_mean_kernel(
    const float* __restrict__ X, const unsigned char* __restrict__ mask,
    float* __restrict__ O, int G, int D)
{
  int b = blockIdx.x;
  const unsigned char* mk = mask + (long)b * G;
  for (int d = threadIdx.x; d < D; d += 256) {
    float s = 0.f, c = 0.f;
    for (int g = 0; g < G; ++g) {
      if (!mk[g]) { s += X[((long)b * G + g) * D + d]; c += 1.f; }
    }
    O[(long)b * D + d] = s / c;
  }
}

// final DN head: coef softmax -> score -> 1->32 relu -> 32->1 sigmoid
__global__ void dn_kernel(
    const float* __restrict__ overall, const float* __restrict__ coef,
    const float* __restrict__ w1, const float* __restrict__ b1,
    const float* __restrict__ w2, const float* __restrict__ b2,
    float* __restrict__ out)
{
  int b = threadIdx.x;
  if (b >= Bb) return;
  float c0 = coef[0], c1 = coef[1];
  float m  = fmaxf(c0, c1);
  float e0 = __expf(c0 - m), e1 = __expf(c1 - m);
  float inv = 1.f / (e0 + e1);
  float score = overall[b * 2 + 0] * e0 * inv + overall[b * 2 + 1] * e1 * inv;
  float acc = b2[0];
  #pragma unroll
  for (int j = 0; j < 32; ++j) {
    float hv = w1[j] * score + b1[j];
    hv = hv > 0.f ? hv : 0.f;
    acc += w2[j] * hv;
  }
  out[b] = 1.f / (1.f + __expf(-acc));
}

// ---------------------------------------------------------------------------
// Host side
// ---------------------------------------------------------------------------
static inline void launch_gemm_abt(hipStream_t st, const float* A, const float* W,
                                   const float* bias, float* C,
                                   int M, int N, int K, int lda, int ldw, int ldc,
                                   float alpha, int accum, int act,
                                   int Z = 1, int zshift = 0,
                                   long sA0 = 0, long sA1 = 0, long sW0 = 0,
                                   long sW1 = 0, long sC0 = 0, long sC1 = 0)
{
  dim3 grid((M + 63) / 64, (N + 127) / 128, Z);
  gemm_abt_wmma<<<grid, 256, 0, st>>>(A, W, bias, C, M, N, K, lda, ldw, ldc,
                                      sA0, sA1, sW0, sW1, sC0, sC1, zshift,
                                      alpha, accum, act);
}

extern "C" void kernel_launch(void* const* d_in, const int* in_sizes, int n_in,
                              void* d_out, int out_size, void* d_ws, size_t ws_size,
                              hipStream_t stream)
{
  (void)in_sizes; (void)n_in; (void)out_size; (void)ws_size;

  const float* glom_embs = (const float*)d_in[0];                 // [8,256,768]
  const float* tile_embs = (const float*)d_in[1];                 // [8,2048,1536]
  const float* coords    = (const float*)d_in[2];                 // [8,2048,2]
  const unsigned char* glom_mask = (const unsigned char*)d_in[3]; // [8,256]
  const unsigned char* tile_mask = (const unsigned char*)d_in[4]; // [8,2048]

  // params flattened jax-tree style (alphabetical keys; lin: b,w; ln: b,g)
  const float* al_k_b = (const float*)d_in[5];   const float* al_k_w = (const float*)d_in[6];
  const float* al_o_b = (const float*)d_in[7];   const float* al_o_w = (const float*)d_in[8];
  const float* al_q_b = (const float*)d_in[9];   const float* al_q_w = (const float*)d_in[10];
  const float* al_v_b = (const float*)d_in[11];  const float* al_v_w = (const float*)d_in[12];
  const float* at_k_b = (const float*)d_in[13];  const float* at_k_w = (const float*)d_in[14];
  const float* at_o_b = (const float*)d_in[15];  const float* at_o_w = (const float*)d_in[16];
  const float* at_q_b = (const float*)d_in[17];  const float* at_q_w = (const float*)d_in[18];
  const float* at_v_b = (const float*)d_in[19];  const float* at_v_w = (const float*)d_in[20];
  const float* coef   = (const float*)d_in[21];
  const float* dn1_b  = (const float*)d_in[22];  const float* dn1_w  = (const float*)d_in[23];
  const float* dn2_b  = (const float*)d_in[24];  const float* dn2_w  = (const float*)d_in[25];
  const float* f1a_b  = (const float*)d_in[26];  const float* f1a_w  = (const float*)d_in[27];
  const float* f1b_b  = (const float*)d_in[28];  const float* f1b_w  = (const float*)d_in[29];
  const float* f2a_b  = (const float*)d_in[30];  const float* f2a_w  = (const float*)d_in[31];
  const float* f2b_b  = (const float*)d_in[32];  const float* f2b_w  = (const float*)d_in[33];
  const float* fuse_b = (const float*)d_in[34];  const float* fuse_w = (const float*)d_in[35];
  const float* fln_b  = (const float*)d_in[36];  const float* fln_g  = (const float*)d_in[37];
  const float* gc1_b  = (const float*)d_in[38];  const float* gc1_w  = (const float*)d_in[39];
  const float* gc2_b  = (const float*)d_in[40];  const float* gc2_w  = (const float*)d_in[41];
  const float* les1_b = (const float*)d_in[42];  const float* les1_w = (const float*)d_in[43];
  const float* les2_b = (const float*)d_in[44];  const float* les2_w = (const float*)d_in[45];
  const float* les_emb= (const float*)d_in[46];
  const float* lln1_b = (const float*)d_in[47];  const float* lln1_g = (const float*)d_in[48];
  const float* lln2_b = (const float*)d_in[49];  const float* lln2_g = (const float*)d_in[50];
  const float* n1_b   = (const float*)d_in[51];  const float* n1_g   = (const float*)d_in[52];
  const float* n2_b   = (const float*)d_in[53];  const float* n2_g   = (const float*)d_in[54];
  const float* n3_b   = (const float*)d_in[55];  const float* n3_g   = (const float*)d_in[56];
  const float* n4_b   = (const float*)d_in[57];  const float* n4_g   = (const float*)d_in[58];
  const float* pc1_b  = (const float*)d_in[59];  const float* pc1_w  = (const float*)d_in[60];
  const float* pc2_b  = (const float*)d_in[61];  const float* pc2_w  = (const float*)d_in[62];
  const float* pool_b = (const float*)d_in[63];  const float* pool_w = (const float*)d_in[64];
  const float* pos1_b = (const float*)d_in[65];  const float* pos1_w = (const float*)d_in[66];
  const float* pos2_b = (const float*)d_in[67];  const float* pos2_w = (const float*)d_in[68];
  const float* pos3_b = (const float*)d_in[69];  const float* pos3_w = (const float*)d_in[70];
  const float* pln_b  = (const float*)d_in[71];  const float* pln_g  = (const float*)d_in[72];
  const float* rel_b  = (const float*)d_in[73];  const float* rel_w  = (const float*)d_in[74];

  float* out = (float*)d_out;     // [0..4095] glom logits, [4096..4111] patient,
                                  // [4112..4127] overall, [4128..4135] dn_prob
  float* ws = (float*)d_ws;

  const long BT = (long)Bb * Tt;      // 16384
  const long BG = (long)Bb * Gg;      // 2048

  // ---- workspace layout (floats) ----
  const long WS_POS   = 0;                          // 16384*1536
  const long WS_TILES = WS_POS   + BT * Dt;         // 16384*1536
  const long WS_K     = WS_TILES + BT * Dt;         // 16384*768
  const long WS_V     = WS_K     + BT * Dg;         // 16384*768
  const long WS_Q     = WS_V     + BT * Dg;         // 2048*768
  const long WS_S     = WS_Q     + BG * Dg;         // 64*256*2048

  // aliases: h1/h2 live in (not-yet-used) S region
  float* h1    = ws + WS_S;                         // 16384*256
  float* h2    = ws + WS_S + BT * 256;              // 16384*512
  float* pos   = ws + WS_POS;
  float* tiles = ws + WS_TILES;
  float* Kbuf  = ws + WS_K;
  float* Vbuf  = ws + WS_V;
  float* Qbuf  = ws + WS_Q;
  float* Sbuf  = ws + WS_S;

  // glom-side small buffers reuse the (dead after fuse) pos region
  float* P = ws + WS_POS;
  float* Obuf  = P;                    // 2048*768  attn output (pre at_o)
  float* gt    = P + 1 * BG * Dg;      // 2048*768
  float* g1    = P + 2 * BG * Dg;
  float* hid   = P + 3 * BG * Dg;      // 2048*1024
  float* fout  = P + 3 * BG * Dg + BG * 1024;
  float* g1b   = fout + BG * Dg;
  float* Qal   = g1b  + BG * Dg;
  float* glc   = Qal  + BG * Dg;       // lesion-attn concat out
  float* gl    = glc  + BG * Dg;
  float* g2    = gl   + BG * Dg;
  float* gf    = g2   + BG * Dg;
  float* gc    = gf   + BG * Dg;       // 2048*384
  float* relev = gc   + BG * 384;      // 2048*2
  float* le1   = relev + BG * 2;       // 2*768
  float* le2   = le1 + 2 * Dg;
  float* Kal   = le2 + 2 * Dg;
  float* Val   = Kal + 2 * Dg;
  float* pooled= Val + 2 * Dg;         // 8*768
  float* poolR = pooled + Bb * Dg;
  float* pcls1o= poolR  + Bb * Dg;     // 8*384

  const float attn_scale = 0.1020620726f;  // 1/sqrt(96)

  // ===== positional encoding MLP =====
  pos1_kernel<<<(BT * 256) / 256, 256, 0, stream>>>(coords, pos1_w, pos1_b, h1);
  launch_gemm_abt(stream, h1, pos2_w, pos2_b, h2, (int)BT, 512, 256, 256, 256, 512,
                  1.f, 0, /*relu*/1);
  launch_gemm_abt(stream, h2, pos3_w, pos3_b, pos, (int)BT, Dt, 512, 512, 512, Dt,
                  1.f, 0, 0);
  ln_kernel<<<(int)BT, 256, 0, stream>>>(pos, nullptr, pln_g, pln_b, pos, Dt, 0);

  // ===== tile fusion: concat handled as two accumulating GEMMs =====
  launch_gemm_abt(stream, tile_embs, fuse_w,        fuse_b,  tiles,
                  (int)BT, Dt, Dt, Dt, 2 * Dt, Dt, 1.f, 0, 0);
  launch_gemm_abt(stream, pos,       fuse_w + Dt,   nullptr, tiles,
                  (int)BT, Dt, Dt, Dt, 2 * Dt, Dt, 1.f, 1, 0);
  ln_kernel<<<(int)BT, 256, 0, stream>>>(tiles, nullptr, fln_g, fln_b, tiles, Dt, /*relu*/1);

  // ===== glom <- tile cross attention =====
  launch_gemm_abt(stream, tiles,     at_k_w, at_k_b, Kbuf, (int)BT, Dg, Dt, Dt, Dt, Dg, 1.f, 0, 0);
  launch_gemm_abt(stream, tiles,     at_v_w, at_v_b, Vbuf, (int)BT, Dg, Dt, Dt, Dt, Dg, 1.f, 0, 0);
  launch_gemm_abt(stream, glom_embs, at_q_w, at_q_b, Qbuf, (int)BG, Dg, Dg, Dg, Dg, Dg, 1.f, 0, 0);

  // scores[z=b*8+h][256][2048] = scale * Q_h K_h^T   (batched, zshift=3)
  launch_gemm_abt(stream, Qbuf, Kbuf, nullptr, Sbuf,
                  Gg, Tt, Dh, Dg, Dg, Tt, attn_scale, 0, 0,
                  /*Z*/Bb * Hh, /*zshift*/3,
                  (long)Gg * Dg, Dh,           // A strides (per b, per h)
                  (long)Tt * Dg, Dh,           // W strides
                  8L * Gg * Tt, (long)Gg * Tt);// C strides
  softmax_mask_kernel<<<(int)(BG * Hh), 256, 0, stream>>>(Sbuf, tile_mask, Tt, Hh * Gg);

  // O_h = P V_h (batched gemm_ab)
  {
    dim3 grid((Gg + 63) / 64, (Dh + 127) / 128, Bb * Hh);
    gemm_ab_wmma<<<grid, 256, 0, stream>>>(Sbuf, Vbuf, Obuf,
        Gg, Dh, Tt, Tt, Dg, Dg,
        8L * Gg * Tt, (long)Gg * Tt,
        (long)Tt * Dg, Dh,
        (long)Gg * Dg, Dh, 3);
  }
  launch_gemm_abt(stream, Obuf, at_o_w, at_o_b, gt, (int)BG, Dg, Dg, Dg, Dg, Dg, 1.f, 0, 0);
  ln_kernel<<<(int)BG, 256, 0, stream>>>(gt, glom_embs, n1_g, n1_b, g1, Dg, 0);

  // FFN1 + norm2
  launch_gemm_abt(stream, g1,  f1a_w, f1a_b, hid,  (int)BG, 1024, Dg, Dg, Dg, 1024, 1.f, 0, 1);
  launch_gemm_abt(stream, hid, f1b_w, f1b_b, fout, (int)BG, Dg, 1024, 1024, 1024, Dg, 1.f, 0, 0);
  ln_kernel<<<(int)BG, 256, 0, stream>>>(fout, g1, n2_g, n2_b, g1b, Dg, 0);

  // ===== lesion encoder (M=2, shared across batch) =====
  linear_small_kernel<<<(2 * Dg + 255) / 256, 256, 0, stream>>>(les_emb, les1_w, les1_b, le1, 2, Dg, Dg, 0);
  ln_kernel<<<2, 256, 0, stream>>>(le1, nullptr, lln1_g, lln1_b, le1, Dg, /*relu*/1);
  linear_small_kernel<<<(2 * Dg + 255) / 256, 256, 0, stream>>>(le1, les2_w, les2_b, le2, 2, Dg, Dg, 0);
  ln_kernel<<<2, 256, 0, stream>>>(le2, nullptr, lln2_g, lln2_b, le2, Dg, 0);
  linear_small_kernel<<<(2 * Dg + 255) / 256, 256, 0, stream>>>(le2, al_k_w, al_k_b, Kal, 2, Dg, Dg, 0);
  linear_small_kernel<<<(2 * Dg + 255) / 256, 256, 0, stream>>>(le2, al_v_w, al_v_b, Val, 2, Dg, Dg, 0);

  // ===== glom <- lesion cross attention =====
  launch_gemm_abt(stream, g1b, al_q_w, al_q_b, Qal, (int)BG, Dg, Dg, Dg, Dg, Dg, 1.f, 0, 0);
  les_attn_kernel<<<(int)(BG * Hh), 32, 0, stream>>>(Qal, Kal, Val, glc);
  launch_gemm_abt(stream, glc, al_o_w, al_o_b, gl, (int)BG, Dg, Dg, Dg, Dg, Dg, 1.f, 0, 0);
  ln_kernel<<<(int)BG, 256, 0, stream>>>(gl, g1b, n3_g, n3_b, g2, Dg, 0);

  // FFN2 + norm4 -> glom_final
  launch_gemm_abt(stream, g2,  f2a_w, f2a_b, hid,  (int)BG, 1024, Dg, Dg, Dg, 1024, 1.f, 0, 1);
  launch_gemm_abt(stream, hid, f2b_w, f2b_b, fout, (int)BG, Dg, 1024, 1024, 1024, Dg, 1.f, 0, 0);
  ln_kernel<<<(int)BG, 256, 0, stream>>>(fout, g2, n4_g, n4_b, gf, Dg, 0);

  // ===== heads =====
  // glom lesion logits -> out[0..4095]
  launch_gemm_abt(stream, gf, gc1_w, gc1_b, gc, (int)BG, 384, Dg, Dg, Dg, 384, 1.f, 0, 1);
  linear_small_kernel<<<(int)((BG * 2 + 255) / 256), 256, 0, stream>>>(gc, gc2_w, gc2_b, out, (int)BG, 2, 384, 0);

  // relev = sigmoid(linear(gf))
  linear_small_kernel<<<(int)((BG * 2 + 255) / 256), 256, 0, stream>>>(gf, rel_w, rel_b, relev, (int)BG, 2, Dg, 2);

  // pooled mean + patient classifier -> out[4096..4111]
  masked_mean_kernel<<<Bb, 256, 0, stream>>>(gf, glom_mask, pooled, Gg, Dg);
  linear_small_kernel<<<(Bb * Dg + 255) / 256, 256, 0, stream>>>(pooled, pool_w, pool_b, poolR, Bb, Dg, Dg, 1);
  linear_small_kernel<<<(Bb * 384 + 255) / 256, 256, 0, stream>>>(poolR, pc1_w, pc1_b, pcls1o, Bb, 384, Dg, 1);
  linear_small_kernel<<<1, 256, 0, stream>>>(pcls1o, pc2_w, pc2_b, out + 4096, Bb, 2, 384, 2);

  // overall -> out[4112..4127]
  masked_mean_kernel<<<Bb, 256, 0, stream>>>(relev, glom_mask, out + 4112, Gg, 2);

  // dn_prob -> out[4128..4135]
  dn_kernel<<<1, 32, 0, stream>>>(out + 4112, coef, dn1_w, dn1_b, dn2_w, dn2_b, out + 4128);
}
